// SegmentEmbed_74509092651250
// MI455X (gfx1250) — matrix operand order repaired and checked by reading, not compiled, not run on hardware
//
#include <hip/hip_runtime.h>

typedef __attribute__((ext_vector_type(16))) _Float16 v16h;
typedef __attribute__((ext_vector_type(8)))  float    v8f;

// ---- probe for CDNA5 async global->LDS copy builtins (guarded: build stays
// green on toolchains that don't expose them; falls back to reg-staged copy)
#if defined(__has_builtin)
#  if __has_builtin(__builtin_amdgcn_global_load_async_to_lds_b128)
#    define HAVE_ASYNC_COPY 1
#  endif
#endif
#ifndef HAVE_ASYNC_COPY
#  define HAVE_ASYNC_COPY 0
#endif

// Problem constants (match reference)
#define BB   8
#define CI   3
#define HH   224
#define WW   224
#define SS   512
#define CM   128
#define DD   768
#define KK   16
#define KDIM (CI * KK * KK)      // 768  (im2col reduction dim for conv)
#define WT   (WW / 16)           // 14 pixel-tiles per row
#define NPIX (BB * HH * WW)      // 401408
#define NSUM (BB * SS * CM)      // 524288 floats (2 MB)
#define NCNT (BB * SS)           // 4096
#define WFRAG_CONV ((KDIM / 32) * 8 * 32 * 16)   // 98304 halves = conv_w reshaped
#define WFRAG_PROJ ((CM / 32) * 48 * 32 * 16)    // 98304 halves = proj_w reshaped

#define TILES_TOTAL (BB * HH * WT)               // 25088 pixel tiles
#define CONV_BLOCKS 392
#define CONV_WAVES  8
#define CONV_ITERS  (TILES_TOTAL / (CONV_BLOCKS * CONV_WAVES))  // 8
#define PATCH_ELEMS (CI * 16 * 31)               // 1488 halves per tile patch
#define CONV_SMEM_BYTES (WFRAG_CONV * 2 + CONV_WAVES * PATCH_ELEMS * 2)  // ~215 KB

// ---------------------------------------------------------------- zero init
__global__ void k_zero(float* sums, int* counts) {
    int i = blockIdx.x * blockDim.x + threadIdx.x;
    if (i < NSUM) sums[i] = 0.0f;
    if (i < NCNT) counts[i] = 0;
}

// ------------------------------------------------------- per-(b,seg) counts
__global__ void k_count(const int* __restrict__ seg, int* __restrict__ counts) {
    int p = blockIdx.x * blockDim.x + threadIdx.x;
    if (p >= NPIX) return;
    int b = p / (HH * WW);
    atomicAdd(&counts[b * SS + seg[p]], 1);
}

// ---------------- pre-swizzle weights into exact WMMA B-fragment lane layout
// B 32x16 f16 fragment: lane l supplies N = l&15, K = (l>>4)*8 + (i&7) + (i>>3)*16  (i=0..15)
__global__ void k_prep_weights(const float* __restrict__ conv_w,
                               const float* __restrict__ proj_w,
                               _Float16* __restrict__ wfA,
                               _Float16* __restrict__ wfP) {
    int id = blockIdx.x * blockDim.x + threadIdx.x;
    if (id >= WFRAG_CONV + WFRAG_PROJ) return;
    int which = (id >= WFRAG_CONV);
    int t = which ? (id - WFRAG_CONV) : id;
    int i    = t & 15;
    int lane = (t >> 4) & 31;
    int frag = t >> 9;                      // 512 halves per fragment
    int klocal = ((lane >> 4) << 3) + (i & 7) + ((i >> 3) << 4);    // 0..31
    if (!which) {                           // conv_w [CM][KDIM] row-major
        int kk = frag >> 3, nt = frag & 7;  // 24 K-steps x 8 N-tiles
        int n = nt * 16 + (lane & 15);
        int k = kk * 32 + klocal;
        wfA[t] = (_Float16)conv_w[n * KDIM + k];
    } else {                                // proj_w [DD][CM] row-major
        int kk = frag / 48, nt = frag % 48; // 4 K-steps x 48 N-tiles
        int n = nt * 16 + (lane & 15);
        int k = kk * 32 + klocal;
        wfP[t] = (_Float16)proj_w[n * CM + k];
    }
}

// ------------------- fused conv(im2col WMMA) + bias + ReLU + pos + scatter
__global__ __launch_bounds__(256)
void k_conv_scatter(const float* __restrict__ x,
                    const int*   __restrict__ seg,
                    const float* __restrict__ conv_b,
                    const float* __restrict__ pos,
                    const _Float16* __restrict__ wfA,
                    float* __restrict__ sums) {
    extern __shared__ __align__(16) char smem_raw[];
    _Float16* wlds  = (_Float16*)smem_raw;            // WFRAG_CONV halves
    _Float16* patch = wlds + WFRAG_CONV;              // 8 * PATCH_ELEMS halves

    const int tid  = threadIdx.x;
    const int lane = tid & 31;
    const int wid  = tid >> 5;

    // ---- stage all conv weight fragments into LDS ----
#if HAVE_ASYNC_COPY
    {
        typedef int av4i __attribute__((vector_size(16)));
        typedef __attribute__((address_space(3))) av4i* as3_av4i;
        av4i* gsrc = (av4i*)(size_t)wfA;   // strip const, retype to int4
        // generic LDS address: low 32 bits are the LDS byte offset (aperture)
        as3_av4i ldsv = (as3_av4i)(unsigned int)(size_t)(void*)wlds;
        for (int j = tid; j < (WFRAG_CONV * 2) / 16; j += 256)
            __builtin_amdgcn_global_load_async_to_lds_b128(gsrc + j, ldsv + j, 0, 0);
#  if defined(__has_builtin) && __has_builtin(__builtin_amdgcn_s_wait_asynccnt)
        __builtin_amdgcn_s_wait_asynccnt(0);
#  else
        asm volatile("s_wait_asynccnt 0x0" ::: "memory");
#  endif
    }
#else
    {
        const uint4* src = (const uint4*)wfA;
        uint4*       dst = (uint4*)wlds;
        for (int j = tid; j < (WFRAG_CONV * 2) / 16; j += 256) dst[j] = src[j];
    }
#endif
    __syncthreads();

    _Float16* mypatch = patch + wid * PATCH_ELEMS;
    const int mA  = lane & 15;           // A-matrix row this lane supplies
    const int kb8 = (lane >> 4) * 8;     // A K-group base (ISA 16-bit A layout)
    const int nlo = lane & 15;           // C/D: N = lane&15, M = r + 8*(lane>>4)
    const int hgp = lane >> 4;
    const float sc = 64.0f / 224.0f;

    for (int iter = 0; iter < CONV_ITERS; ++iter) {
        int t   = blockIdx.x * CONV_WAVES + wid + iter * (CONV_BLOCKS * CONV_WAVES);
        int b   = t / (HH * WT);
        int rem = t % (HH * WT);
        int h   = rem / WT;
        int w0  = (rem % WT) * 16;
        const float* xb = x + (size_t)b * CI * HH * WW;

        __syncthreads();   // uniform: previous-iter patch readers are done
        // ---- stage zero-padded input patch (SAME pad: lo=7, hi=8) ----
        for (int e = lane; e < PATCH_ELEMS; e += 32) {
            int ci = e / (16 * 31);
            int r2 = e % (16 * 31);
            int rr = r2 / 31, cc = r2 % 31;
            int hi = h - 7 + rr, wi = w0 - 7 + cc;
            float v = 0.0f;
            if ((unsigned)hi < (unsigned)HH && (unsigned)wi < (unsigned)WW)
                v = xb[(ci * HH + hi) * WW + wi];
            mypatch[e] = (_Float16)v;
        }
        __syncthreads();   // uniform: patches visible

        v8f acc[8];
#pragma unroll
        for (int nt = 0; nt < 8; ++nt) acc[nt] = (v8f){0, 0, 0, 0, 0, 0, 0, 0};

        // one-fragment lookahead on B so each ds_load has a WMMA of slack
        v16h bcur = *(const v16h*)(wlds + (size_t)lane * 16);
        for (int kk = 0; kk < KDIM / 32; ++kk) {
            // ---- A fragment from LDS patch (2x ds_load_b128) ----
            v16h a;
#pragma unroll
            for (int i = 0; i < 16; ++i) {
                int k  = kk * 32 + kb8 + (i & 7) + ((i >> 3) << 4);
                int ci = k >> 8;             // k / 256
                int kh = (k >> 4) & 15;
                int kw = k & 15;
                a[i] = mypatch[ci * (16 * 31) + kh * 31 + mA + kw];
            }
#pragma unroll
            for (int nt = 0; nt < 8; ++nt) {
                v16h bnext = bcur;
                int fid = kk * 8 + nt + 1;           // next fragment id
                if (fid < (KDIM / 32) * 8)
                    bnext = *(const v16h*)(wlds + (size_t)(fid * 32 + lane) * 16);
                acc[nt] = __builtin_amdgcn_wmma_f32_16x16x32_f16(
                    false, a, false, bcur, (short)0, acc[nt], false, false);
                bcur = bnext;
            }
        }

        // ---- epilogue: bias + ReLU + pos bilinear (64->224) + scatter ----
        float cb[8];
#pragma unroll
        for (int nt = 0; nt < 8; ++nt) cb[nt] = conv_b[nt * 16 + nlo];

        float iy = (h + 0.5f) * sc - 0.5f;
        int   y0 = (int)floorf(iy);
        float fy = iy - (float)y0;
        int y0c = min(max(y0, 0), 63), y1c = min(max(y0 + 1, 0), 63);
        const int* segrow = seg + ((size_t)b * HH + h) * WW;

#pragma unroll
        for (int r = 0; r < 8; ++r) {
            int m  = r + 8 * hgp;
            int wp = w0 + m;
            float ix = (wp + 0.5f) * sc - 0.5f;
            int   x0 = (int)floorf(ix);
            float fx = ix - (float)x0;
            int x0c = min(max(x0, 0), 63), x1c = min(max(x0 + 1, 0), 63);
            int sidx = segrow[wp];
            float* srow = sums + (size_t)(b * SS + sidx) * CM;
#pragma unroll
            for (int nt = 0; nt < 8; ++nt) {
                int ch = nt * 16 + nlo;
                float v = acc[nt][r] + cb[nt];
                v = v > 0.0f ? v : 0.0f;
                const float* pc = pos + (size_t)ch * 64 * 64;
                float p00 = pc[y0c * 64 + x0c], p01 = pc[y0c * 64 + x1c];
                float p10 = pc[y1c * 64 + x0c], p11 = pc[y1c * 64 + x1c];
                float pv = (p00 * (1.f - fx) + p01 * fx) * (1.f - fy)
                         + (p10 * (1.f - fx) + p11 * fx) * fy;
                atomicAdd(&srow[ch], v + pv);
            }
        }
    }
}

// -------- projection GEMM: out[m,n] = (sums[m,:]/cnt[m]) @ proj_w^T + bias
// M = B*S = 4096, N = D = 768, K = CM = 128 -> WMMA f16, 4 K-steps per tile.
__global__ __launch_bounds__(256)
void k_proj(const float* __restrict__ sums,
            const int*   __restrict__ counts,
            const _Float16* __restrict__ wfP,
            const float* __restrict__ proj_b,
            float* __restrict__ out) {
    const int lane = threadIdx.x & 31;
    const int wave = blockIdx.x * (blockDim.x >> 5) + (threadIdx.x >> 5);
    int mt = wave / 48;                      // 256 M-tiles
    int nt = wave % 48;                      // 48 N-tiles

    const int mA   = mt * 16 + (lane & 15);
    const float inv = 1.0f / fmaxf((float)counts[mA], 1.0f);
    const float* arow = sums + (size_t)mA * CM;
    const int kb8 = (lane >> 4) * 8;

    v8f acc = (v8f){0, 0, 0, 0, 0, 0, 0, 0};
#pragma unroll
    for (int kk = 0; kk < CM / 32; ++kk) {
        v16h a;
#pragma unroll
        for (int i = 0; i < 16; ++i) {
            int k = kk * 32 + kb8 + (i & 7) + ((i >> 3) << 4);
            a[i] = (_Float16)(arow[k] * inv);
        }
        v16h bf = *(const v16h*)(wfP + (size_t)((kk * 48 + nt) * 32 + lane) * 16);
        acc = __builtin_amdgcn_wmma_f32_16x16x32_f16(
            false, a, false, bf, (short)0, acc, false, false);
    }

    const int nlo = lane & 15, hgp = lane >> 4;
    int n = nt * 16 + nlo;
    float bias = proj_b[n];
#pragma unroll
    for (int r = 0; r < 8; ++r) {
        int m = mt * 16 + r + 8 * hgp;
        out[(size_t)m * DD + n] = acc[r] + bias;
    }
}

// ----------------------- second tuple output: segment ids (bit copy)
__global__ void k_copy_seg(const int* __restrict__ seg, int* __restrict__ tail) {
    int i = blockIdx.x * blockDim.x + threadIdx.x;
    if (i < NPIX) tail[i] = seg[i];
}

extern "C" void kernel_launch(void* const* d_in, const int* in_sizes, int n_in,
                              void* d_out, int out_size, void* d_ws, size_t ws_size,
                              hipStream_t stream) {
    const float* x      = (const float*)d_in[0];
    const int*   seg    = (const int*)  d_in[1];
    const float* conv_w = (const float*)d_in[2];
    const float* conv_b = (const float*)d_in[3];
    const float* pos    = (const float*)d_in[4];
    const float* proj_w = (const float*)d_in[5];
    const float* proj_b = (const float*)d_in[6];
    // d_in[7] = n_segments scalar (compile-time SS here)
    float* out = (float*)d_out;

    // workspace carve-up (aligned): sums | counts | conv B-frags | proj B-frags
    char* ws = (char*)d_ws;
    float*     sums   = (float*)ws;                                   // 2 MB
    int*       counts = (int*)(ws + (size_t)NSUM * 4);                // 16 KB
    _Float16*  wfA    = (_Float16*)(ws + (size_t)NSUM * 4 + 64 * 1024);
    _Float16*  wfP    = wfA + WFRAG_CONV;

    k_zero<<<(NSUM + 255) / 256, 256, 0, stream>>>(sums, counts);
    k_count<<<(NPIX + 255) / 256, 256, 0, stream>>>(seg, counts);
    k_prep_weights<<<(WFRAG_CONV + WFRAG_PROJ + 255) / 256, 256, 0, stream>>>(
        conv_w, proj_w, wfA, wfP);

    // 392 blocks x 8 waves x 8 tile-iterations = 25088 tiles exactly
    k_conv_scatter<<<CONV_BLOCKS, 256, CONV_SMEM_BYTES, stream>>>(
        x, seg, conv_b, pos, wfA, sums);

    // 256*48 = 12288 GEMM tiles, 8 waves per block -> 1536 blocks exactly
    k_proj<<<(256 * 48) / 8, 256, 0, stream>>>(sums, counts, wfP, proj_b, out);

    k_copy_seg<<<(NPIX + 255) / 256, 256, 0, stream>>>(
        seg, (int*)(out + (size_t)BB * SS * DD));
}
